// TripletLoss_41583873360461
// MI455X (gfx1250) — 8-bit matrix operand order repaired and checked by reading, compiled, not run on hardware
//
#include <hip/hip_runtime.h>
#include <hip/hip_bf16.h>

#define BN 320
#define DD 128
#define CC 80
#define CPAD 128
#define NANCH 30
#define MARGIN 0.5f

typedef float v2f __attribute__((ext_vector_type(2)));
typedef float v8f __attribute__((ext_vector_type(8)));
typedef int   v8i __attribute__((ext_vector_type(8)));
typedef int   v4i __attribute__((ext_vector_type(4)));
typedef int   v2i __attribute__((ext_vector_type(2)));

__global__ void tl_norms_kernel(const float* __restrict__ src,
                                const float* __restrict__ tgt,
                                float* __restrict__ ns, float* __restrict__ nt) {
    int id = blockIdx.x;
    const float* row = (id < BN) ? (src + (size_t)id * DD)
                                 : (tgt + (size_t)(id - BN) * DD);
    int lane = threadIdx.x;
    float s = 0.f;
    #pragma unroll
    for (int k = lane; k < DD; k += 32) { float v = row[k]; s += v * v; }
    #pragma unroll
    for (int o = 16; o > 0; o >>= 1) s += __shfl_xor(s, o, 32);
    if (lane == 0) {
        float r = sqrtf(s);
        if (id < BN) ns[id] = r; else nt[id - BN] = r;
    }
}

__global__ void tl_pack_labels_kernel(const int* __restrict__ lab,
                                      signed char* __restrict__ Lb,
                                      float* __restrict__ rs) {
    int row = blockIdx.x;
    int c   = threadIdx.x;
    int v = (c < CC) ? lab[row * CC + c] : 0;
    Lb[row * CPAD + c] = (signed char)v;
    __shared__ int red[128];
    red[c] = v;
    __syncthreads();
    #pragma unroll
    for (int o = 64; o > 0; o >>= 1) {
        if (c < o) red[c] += red[c + o];
        __syncthreads();
    }
    if (c == 0) rs[row] = (float)red[0];
}

__global__ void tl_fd_wmma_kernel(const float* __restrict__ src,
                                  const float* __restrict__ tgt,
                                  const float* __restrict__ ns,
                                  const float* __restrict__ nt,
                                  float* __restrict__ fd) {
    int row0 = blockIdx.y * 16, col0 = blockIdx.x * 16;
    int lane = threadIdx.x;
    int m  = lane & 15;
    int hi = lane >> 4;
    const float* arow = src + (size_t)(row0 + m) * DD;
    const float* brow = tgt + (size_t)(col0 + m) * DD;
    v8f acc = {};
    #pragma unroll 4
    for (int k0 = 0; k0 < DD; k0 += 4) {
        v2f a = *(const v2f*)(arow + k0 + 2 * hi);
        v2f b = *(const v2f*)(brow + k0 + 2 * hi);
        acc = __builtin_amdgcn_wmma_f32_16x16x4_f32(
            false, a, false, b, (short)0, acc, false, false);
    }
    #pragma unroll
    for (int r = 0; r < 8; ++r) {
        int i = row0 + r + 8 * hi;
        int j = col0 + m;
        float sim = acc[r] / fmaxf(ns[i] * nt[j], 1e-8f);
        fd[i * BN + j] = fmaxf(1.0f - sim, 0.0f);
    }
}

__global__ void tl_gram_iu8_kernel(const signed char* __restrict__ Lb,
                                   const float* __restrict__ rs,
                                   unsigned char* __restrict__ posM,
                                   unsigned char* __restrict__ negM) {
    int row0 = blockIdx.y * 16, col0 = blockIdx.x * 16;
    int lane = threadIdx.x;
    int m  = lane & 15;
    int hi = lane >> 4;
    const signed char* arow = Lb + (size_t)(row0 + m) * CPAD;
    const signed char* brow = Lb + (size_t)(col0 + m) * CPAD;
    v8i acc = {};
    #pragma unroll
    for (int t = 0; t < 2; ++t) {
        int kb = t * 64;
        v8i a, b;
        #pragma unroll
        for (int j = 0; j < 4; ++j) {
            v2i ch = *(const v2i*)(arow + kb + 16 * j + 8 * hi);
            a[2 * j]     = ch.x;
            a[2 * j + 1] = ch.y;
        }
        #pragma unroll
        for (int j = 0; j < 4; ++j) {
            v2i cb = *(const v2i*)(brow + kb + 16 * j + 8 * hi);
            b[2 * j]     = cb.x;
            b[2 * j + 1] = cb.y;
        }
        acc = __builtin_amdgcn_wmma_i32_16x16x64_iu8(
            false, a, false, b, acc, false, false);
    }
    #pragma unroll
    for (int r = 0; r < 8; ++r) {
        int i = row0 + r + 8 * hi;
        int j = col0 + m;
        float dot   = (float)acc[r];
        float ratio = dot / sqrtf(rs[i] * rs[j]);
        posM[i * BN + j] = (ratio >= 0.8f) ? 1 : 0;
        negM[i * BN + j] = (ratio <= 0.5f) ? 1 : 0;
    }
}

__global__ void tl_div_kernel(const float* __restrict__ fd, float* __restrict__ divr) {
    int a = blockIdx.x, lane = threadIdx.x;
    float s = 0.f;
    #pragma unroll
    for (int j = lane; j < BN; j += 32) s += fd[a * BN + j];
    #pragma unroll
    for (int o = 16; o > 0; o >>= 1) s += __shfl_xor(s, o, 32);
    if (lane == 0) divr[a] = s;
}

__global__ void tl_anchor_kernel(const float* __restrict__ divr, int* __restrict__ anchor) {
    int a = threadIdx.x;
    float d = divr[a];
    int rank = 0;
    for (int j = 0; j < BN; ++j) {
        float dj = divr[j];
        rank += (dj > d) || (dj == d && j < a);
    }
    anchor[a] = (rank < NANCH) ? 1 : 0;
}

__global__ void tl_loss_kernel(const float* __restrict__ fd,
                               const unsigned char* __restrict__ posM,
                               const unsigned char* __restrict__ negM,
                               const int* __restrict__ anchor,
                               float* __restrict__ psum, float* __restrict__ pcnt) {
    int a   = blockIdx.x;
    int tid = threadIdx.x;
    if (!anchor[a]) {
        if (tid == 0) { psum[a] = 0.f; pcnt[a] = 0.f; }
        return;
    }
    __shared__ float fda[BN];
    __shared__ short plist[BN];
    __shared__ short nlist[BN];
    __shared__ int   cnt[2];
    if (tid < 2) cnt[tid] = 0;
    __syncthreads();
    for (int i = tid; i < BN; i += 256) {
        fda[i] = fd[a * BN + i];
        if (i != a) {
            if (posM[a * BN + i]) plist[atomicAdd(&cnt[0], 1)] = (short)i;
            if (negM[a * BN + i]) nlist[atomicAdd(&cnt[1], 1)] = (short)i;
        }
    }
    __syncthreads();
    int np = cnt[0], nn = cnt[1];
    int total = np * nn;
    float s = 0.f;
    for (int idx = tid; idx < total; idx += 256) {
        int p = plist[idx / nn];
        int n = nlist[idx % nn];
        s += fmaxf(fda[p] - fda[n] + MARGIN, 0.0f);
    }
    __shared__ float red[256];
    red[tid] = s;
    __syncthreads();
    #pragma unroll
    for (int o = 128; o > 0; o >>= 1) {
        if (tid < o) red[tid] += red[tid + o];
        __syncthreads();
    }
    if (tid == 0) { psum[a] = red[0]; pcnt[a] = (float)total; }
}

__global__ void tl_finalize_kernel(const float* __restrict__ psum,
                                   const float* __restrict__ pcnt,
                                   float* __restrict__ out) {
    float s = 0.f, c = 0.f;
    for (int a = 0; a < BN; ++a) { s += psum[a]; c += pcnt[a]; }
    out[0] = s / (c + 1e-4f);
}

extern "C" void kernel_launch(void* const* d_in, const int* in_sizes, int n_in,
                              void* d_out, int out_size, void* d_ws, size_t ws_size,
                              hipStream_t stream) {
    const int*   lab = (const int*)d_in[0];
    const float* src = (const float*)d_in[1];
    const float* tgt = (const float*)d_in[2];
    float* out = (float*)d_out;

    char* ws = (char*)d_ws;
    size_t off = 0;
    float* fd   = (float*)(ws + off); off += (size_t)BN * BN * sizeof(float);
    float* ns   = (float*)(ws + off); off += BN * sizeof(float);
    float* nt   = (float*)(ws + off); off += BN * sizeof(float);
    float* rs   = (float*)(ws + off); off += BN * sizeof(float);
    float* divr = (float*)(ws + off); off += BN * sizeof(float);
    int*   anch = (int*)  (ws + off); off += BN * sizeof(int);
    float* psum = (float*)(ws + off); off += BN * sizeof(float);
    float* pcnt = (float*)(ws + off); off += BN * sizeof(float);
    signed char*   Lb   = (signed char*)(ws + off);   off += (size_t)BN * CPAD;
    unsigned char* posM = (unsigned char*)(ws + off); off += (size_t)BN * BN;
    unsigned char* negM = (unsigned char*)(ws + off); off += (size_t)BN * BN;

    tl_norms_kernel<<<2 * BN, 32, 0, stream>>>(src, tgt, ns, nt);
    tl_pack_labels_kernel<<<BN, 128, 0, stream>>>(lab, Lb, rs);

    dim3 tiles(BN / 16, BN / 16);
    tl_fd_wmma_kernel<<<tiles, 32, 0, stream>>>(src, tgt, ns, nt, fd);
    tl_gram_iu8_kernel<<<tiles, 32, 0, stream>>>(Lb, rs, posM, negM);

    tl_div_kernel<<<BN, 32, 0, stream>>>(fd, divr);
    tl_anchor_kernel<<<1, BN, 0, stream>>>(divr, anch);
    tl_loss_kernel<<<BN, 256, 0, stream>>>(fd, posM, negM, anch, psum, pcnt);
    tl_finalize_kernel<<<1, 1, 0, stream>>>(psum, pcnt, out);
}
